// SSC_GT_64295660421369
// MI455X (gfx1250) — matrix operand order repaired and checked by reading, not compiled, not run on hardware
//
#include <hip/hip_runtime.h>
#include <hip/hip_bf16.h>
#include <math.h>

// ---------------------------------------------------------------------------
// Problem constants (match reference)
// ---------------------------------------------------------------------------
#define BATCH 16
#define CCH   256
#define HDIM  96
#define WDIM  96
#define HW    9216          // 96*96
#define NDIM  64            // NODE_DIM
#define MNODE 64            // k == MAX_NODES
#define SHID  64            // SCORE_HIDDEN
#define NTILE 144           // HW / 64 pixel tiles

// ---------------------------------------------------------------------------
// WMMA vector types / helpers (gfx1250, wave32)
// ---------------------------------------------------------------------------
typedef __attribute__((ext_vector_type(16))) _Float16     v16h;
typedef __attribute__((ext_vector_type(8)))  float        v8f;
typedef __attribute__((ext_vector_type(4)))  unsigned int u32x4;

__device__ __forceinline__ v8f vzero8() {
  v8f z = {0.f, 0.f, 0.f, 0.f, 0.f, 0.f, 0.f, 0.f};
  return z;
}

// A-fragment (16x32 f16), source row-major [row][k], row stride ld halves.
// Lanes 0-15 row M=lane, K [0..16); lanes 16-31 same rows, K [16..32)
// (contiguous, matching load_frag_b)  ->  chunk0 = kb + half*16, chunk1 = +8.
__device__ __forceinline__ v16h load_frag_a(const _Float16* base, int ld,
                                            int kb, int lane) {
  const int r  = lane & 15;
  const int h8 = (lane >> 4) * 16;
  const _Float16* row = base + (size_t)r * ld + kb;
  union { v16h v; u32x4 q[2]; } f;
  f.q[0] = *(const u32x4*)(row + h8);
  f.q[1] = *(const u32x4*)(row + 8 + h8);
  return f.v;
}

// B-fragment (32x16 f16), source stored as [n][k] row-major (column n of the
// mathematical KxN B matrix is a contiguous row). Contiguous k (same as A): lanes 0-15 hold
// K=0-15, lanes 16-31 hold K=16-31, column = lane&15 -> one 32B load.
__device__ __forceinline__ v16h load_frag_b(const _Float16* base, int ld,
                                            int kb, int lane) {
  const int n   = lane & 15;
  const int h16 = (lane >> 4) * 16;
  return *(const v16h*)(base + (size_t)n * ld + kb + h16);
}

__device__ __forceinline__ v8f wmma_f16(v16h a, v16h b, v8f c) {
  return __builtin_amdgcn_wmma_f32_16x16x32_f16(
      /*neg_a=*/false, a, /*neg_b=*/false, b,
      /*c_mod=*/(short)0, c, /*reuse_a=*/false, /*reuse_b=*/false);
}

// ---------------------------------------------------------------------------
// K0: f32 -> f16 weight conversion (layouts are already [n][k] row-major)
// ---------------------------------------------------------------------------
__global__ void ssc_k0_cvt(const float* __restrict__ src,
                           _Float16* __restrict__ dst, int n) {
  int i = blockIdx.x * 256 + threadIdx.x;
  if (i < n) dst[i] = (_Float16)src[i];
}

// ---------------------------------------------------------------------------
// K1: fused score-hidden + token projection.
// Grid (NTILE, BATCH), 128 threads (4 waves); each wave owns 16 pixels.
// ---------------------------------------------------------------------------
__global__ __launch_bounds__(128) void ssc_k1_score_proj(
    const float* __restrict__ fmap, const _Float16* __restrict__ ws1h,
    const _Float16* __restrict__ wt2nh, const float* __restrict__ bs1,
    const float* __restrict__ ws2, const float* __restrict__ bs2,
    _Float16* __restrict__ tokens, float* __restrict__ scores) {
  const int tile = blockIdx.x, b = blockIdx.y;
  const int tid = threadIdx.x, lane = tid & 31, w = tid >> 5;
  const int p0 = tile * 64;

  __shared__ _Float16 Af[64 * CCH] __attribute__((aligned(16)));  // [pixel][c]
  __shared__ float hid[64 * SHID];                                // [pixel][o]

  // Stage fmap tile (channel-major -> pixel-major f16). Coalesced over p.
  const float* fm = fmap + (size_t)b * CCH * HW;
  for (int e = tid; e < 64 * CCH; e += 128) {
    int c = e >> 6, p = e & 63;
    Af[p * CCH + c] = (_Float16)fm[(size_t)c * HW + p0 + p];
  }
  __syncthreads();

  const _Float16* Abase = Af + w * 16 * CCH;
  v8f acc_h[4], acc_t[4];
  for (int t = 0; t < 4; ++t) { acc_h[t] = vzero8(); acc_t[t] = vzero8(); }

  // K = 256 -> 8 steps of 32; N = 64 -> 4 tiles for hid and 4 for tokens.
  // Load ALL B fragments of a k-step first so the 8 global b128 clauses issue
  // under one wait, then run the 8 WMMAs back-to-back.
  for (int ks = 0; ks < 8; ++ks) {
    const int kb = ks * 32;
    v16h a = load_frag_a(Abase, CCH, kb, lane);
    v16h b1[4], b2[4];
    for (int t = 0; t < 4; ++t) {
      b1[t] = load_frag_b(ws1h + (size_t)(t * 16) * CCH, CCH, kb, lane);
      b2[t] = load_frag_b(wt2nh + (size_t)(t * 16) * CCH, CCH, kb, lane);
    }
    for (int t = 0; t < 4; ++t) {
      acc_h[t] = wmma_f16(a, b1[t], acc_h[t]);
      acc_t[t] = wmma_f16(a, b2[t], acc_t[t]);
    }
  }

  const int M0 = (lane >> 4) * 8;   // row offset within 16-row tile
  const int N  = lane & 15;         // output column within 16-col tile

  // hid = relu(acc + bias) -> LDS (for score dot)
  for (int t = 0; t < 4; ++t) {
    float bias = bs1[t * 16 + N];
    for (int j = 0; j < 8; ++j) {
      float h = acc_h[t][j] + bias;
      hid[(w * 16 + M0 + j) * SHID + t * 16 + N] = fmaxf(h, 0.f);
    }
  }
  // tokens (no bias / no relu) -> HBM f16, row-major [pixel][d]
  const size_t tokbase = ((size_t)b * HW + p0 + w * 16) * NDIM;
  for (int t = 0; t < 4; ++t)
    for (int j = 0; j < 8; ++j)
      tokens[tokbase + (size_t)(M0 + j) * NDIM + t * 16 + N] =
          (_Float16)acc_t[t][j];
  __syncthreads();

  // score = hid . ws2 + bs2 (one thread per pixel row)
  if (tid < 64) {
    float s = bs2[0];
    const float* hrow = hid + tid * SHID;
    for (int n = 0; n < SHID; ++n) s += hrow[n] * ws2[n];
    scores[(size_t)b * HW + p0 + tid] = s;
  }
}

// ---------------------------------------------------------------------------
// K2: per-batch top-64 by iterative argmax (stable: lowest index on ties).
// Destructive on `scores` (recomputed by K1 every call -> deterministic).
// ---------------------------------------------------------------------------
__global__ __launch_bounds__(256) void ssc_k2_topk(float* __restrict__ scores,
                                                   int* __restrict__ idx) {
  const int b = blockIdx.x, tid = threadIdx.x;
  float* sc = scores + (size_t)b * HW;
  __shared__ float sv[256];
  __shared__ int   si[256];
  for (int it = 0; it < MNODE; ++it) {
    float best = -3.4e38f; int bi = HW;
    for (int p = tid; p < HW; p += 256) {
      float v = sc[p];
      if (v > best || (v == best && p < bi)) { best = v; bi = p; }
    }
    sv[tid] = best; si[tid] = bi;
    __syncthreads();
    for (int s = 128; s > 0; s >>= 1) {
      if (tid < s) {
        float v = sv[tid + s]; int i2 = si[tid + s];
        if (v > sv[tid] || (v == sv[tid] && i2 < si[tid])) {
          sv[tid] = v; si[tid] = i2;
        }
      }
      __syncthreads();
    }
    if (tid == 0) { idx[b * MNODE + it] = si[0]; sc[si[0]] = -3.4e38f; }
    __syncthreads();
  }
}

// ---------------------------------------------------------------------------
// K3: gather + cosine adjacency + 2-layer GCN (64x64 per batch, f32 VALU).
// Emits Hg as f16 in both [m][d] (logits B) and [d][m] (inject B) layouts.
// ---------------------------------------------------------------------------
__global__ __launch_bounds__(256) void ssc_k3_graph(
    const _Float16* __restrict__ tokens, const int* __restrict__ idx,
    const float* __restrict__ wg1, const float* __restrict__ wg2,
    _Float16* __restrict__ Hgh, _Float16* __restrict__ HgTh) {
  const int b = blockIdx.x, tid = threadIdx.x;
  __shared__ float B0[64 * 64];   // H0 -> X -> Hg
  __shared__ float B1[64 * 64];   // A
  __shared__ float B2[64 * 64];   // Y1 -> Y2
  __shared__ float nrm[64];

  const _Float16* tok = tokens + (size_t)b * HW * NDIM;
  const int* id = idx + b * MNODE;

  // gather H0
  for (int e = tid; e < 4096; e += 256) {
    int m = e >> 6, d = e & 63;
    B0[e] = (float)tok[(size_t)id[m] * NDIM + d];
  }
  __syncthreads();
  if (tid < 64) {
    float s = 0.f;
    for (int d = 0; d < 64; ++d) { float v = B0[tid * 64 + d]; s += v * v; }
    nrm[tid] = fmaxf(sqrtf(s), 1e-6f);
  }
  __syncthreads();
  // A = relu(cos) + I
  for (int e = tid; e < 4096; e += 256) {
    int m = e >> 6, n = e & 63;
    float dot = 0.f;
    for (int d = 0; d < 64; ++d) dot += B0[m * 64 + d] * B0[n * 64 + d];
    float a = fmaxf(dot / (nrm[m] * nrm[n]), 0.f);
    if (m == n) a += 1.f;
    B1[e] = a;
  }
  __syncthreads();
  // row-normalize A
  if (tid < 64) {
    float s = 0.f;
    for (int n = 0; n < 64; ++n) s += B1[tid * 64 + n];
    float inv = 1.f / fmaxf(s, 1e-6f);
    for (int n = 0; n < 64; ++n) B1[tid * 64 + n] *= inv;
  }
  __syncthreads();
  // Y1 = H0 @ wg1^T
  for (int e = tid; e < 4096; e += 256) {
    int m = e >> 6, o = e & 63;
    float s = 0.f;
    for (int d = 0; d < 64; ++d) s += B0[m * 64 + d] * wg1[o * 64 + d];
    B2[e] = s;
  }
  __syncthreads();
  // X = relu(A @ Y1)  (into B0; H0 dead)
  for (int e = tid; e < 4096; e += 256) {
    int m = e >> 6, o = e & 63;
    float s = 0.f;
    for (int n = 0; n < 64; ++n) s += B1[m * 64 + n] * B2[n * 64 + o];
    B0[e] = fmaxf(s, 0.f);
  }
  __syncthreads();
  // Y2 = X @ wg2^T  (into B2; Y1 dead)
  for (int e = tid; e < 4096; e += 256) {
    int m = e >> 6, o = e & 63;
    float s = 0.f;
    for (int d = 0; d < 64; ++d) s += B0[m * 64 + d] * wg2[o * 64 + d];
    B2[e] = s;
  }
  __syncthreads();
  // Hg = relu(A @ Y2)  (into B0; X dead)
  for (int e = tid; e < 4096; e += 256) {
    int m = e >> 6, o = e & 63;
    float s = 0.f;
    for (int n = 0; n < 64; ++n) s += B1[m * 64 + n] * B2[n * 64 + o];
    B0[e] = fmaxf(s, 0.f);
  }
  __syncthreads();
  for (int e = tid; e < 4096; e += 256) {
    int m = e >> 6, d = e & 63;
    _Float16 h = (_Float16)B0[m * 64 + d];
    Hgh [(size_t)b * 4096 + m * 64 + d] = h;   // [m][d]
    HgTh[(size_t)b * 4096 + d * 64 + m] = h;   // [d][m]
  }
}

// ---------------------------------------------------------------------------
// K4: fused logits -> softmax -> inject -> back-projection + residual.
// Grid (NTILE, BATCH), 128 threads (4 waves); each wave owns 16 pixels.
// ---------------------------------------------------------------------------
__global__ __launch_bounds__(128) void ssc_k4_attn(
    const float* __restrict__ fmap, const _Float16* __restrict__ tokens,
    const _Float16* __restrict__ Hgh, const _Float16* __restrict__ HgTh,
    const _Float16* __restrict__ wn2th, float* __restrict__ out) {
  const int tile = blockIdx.x, b = blockIdx.y;
  const int tid = threadIdx.x, lane = tid & 31, w = tid >> 5;
  const int p0 = tile * 64 + w * 16;     // first pixel of this wave's tile

  __shared__ float    lg [64 * MNODE];                              // logits
  __shared__ _Float16 at [64 * MNODE] __attribute__((aligned(16))); // attn
  __shared__ _Float16 inj[64 * NDIM ] __attribute__((aligned(16))); // injected

  const _Float16* tok = tokens + ((size_t)b * HW + p0) * NDIM;
  const _Float16* Hg  = Hgh  + (size_t)b * 4096;
  const _Float16* HgT = HgTh + (size_t)b * 4096;
  const int M0 = (lane >> 4) * 8;
  const int N  = lane & 15;
  const int rbase = w * 16;              // row base within 64-row block tile
  const float* fmb = fmap + (size_t)b * CCH * HW;
  float* outb      = out  + (size_t)b * CCH * HW;

  // Warm L2 for the residual rows we will read in stage 4.
  for (int ct = 0; ct < 16; ++ct)
    __builtin_prefetch(fmb + (size_t)(ct * 16 + N) * HW + p0 + M0, 0, 1);

  // ---- stage 1: logits = tokens . Hg^T / sqrt(d) ----
  {
    v16h a0 = load_frag_a(tok, NDIM, 0, lane);
    v16h a1 = load_frag_a(tok, NDIM, 32, lane);
    v16h b0[4], b1[4];
    for (int mt = 0; mt < 4; ++mt) {
      b0[mt] = load_frag_b(Hg + (size_t)(mt * 16) * NDIM, NDIM, 0, lane);
      b1[mt] = load_frag_b(Hg + (size_t)(mt * 16) * NDIM, NDIM, 32, lane);
    }
    for (int mt = 0; mt < 4; ++mt) {
      v8f acc = vzero8();
      acc = wmma_f16(a0, b0[mt], acc);
      acc = wmma_f16(a1, b1[mt], acc);
      for (int j = 0; j < 8; ++j)
        lg[(rbase + M0 + j) * MNODE + mt * 16 + N] = acc[j] * 0.125f;
    }
  }
  __syncthreads();

  // ---- stage 2: softmax over M=64, one thread per pixel row ----
  if (tid < 64) {
    float* row = lg + tid * MNODE;
    float mx = row[0];
    for (int m = 1; m < MNODE; ++m) mx = fmaxf(mx, row[m]);
    float sum = 0.f;
    for (int m = 0; m < MNODE; ++m) { float e = __expf(row[m] - mx); row[m] = e; sum += e; }
    float inv = 1.f / sum;
    for (int m = 0; m < MNODE; ++m) at[tid * MNODE + m] = (_Float16)(row[m] * inv);
  }
  __syncthreads();

  // ---- stage 3: injected = attn @ Hg ----
  {
    v16h a0 = load_frag_a(at + (size_t)rbase * MNODE, MNODE, 0, lane);
    v16h a1 = load_frag_a(at + (size_t)rbase * MNODE, MNODE, 32, lane);
    v16h b0[4], b1[4];
    for (int et = 0; et < 4; ++et) {
      b0[et] = load_frag_b(HgT + (size_t)(et * 16) * NDIM, NDIM, 0, lane);
      b1[et] = load_frag_b(HgT + (size_t)(et * 16) * NDIM, NDIM, 32, lane);
    }
    for (int et = 0; et < 4; ++et) {
      v8f acc = vzero8();
      acc = wmma_f16(a0, b0[et], acc);
      acc = wmma_f16(a1, b1[et], acc);
      for (int j = 0; j < 8; ++j)
        inj[(rbase + M0 + j) * NDIM + et * 16 + N] = (_Float16)acc[j];
    }
  }
  __syncthreads();

  // ---- stage 4: back = injected @ wn2t^T ; out = fmap + back ----
  // D-fragment: lane holds 8 consecutive pixels (M=j+M0) of channel c.
  {
    v16h a0 = load_frag_a(inj + (size_t)rbase * NDIM, NDIM, 0, lane);
    v16h a1 = load_frag_a(inj + (size_t)rbase * NDIM, NDIM, 32, lane);
    for (int ct = 0; ct < 16; ++ct) {
      v16h bb0 = load_frag_b(wn2th + (size_t)(ct * 16) * NDIM, NDIM, 0, lane);
      v16h bb1 = load_frag_b(wn2th + (size_t)(ct * 16) * NDIM, NDIM, 32, lane);
      v8f acc = vzero8();
      acc = wmma_f16(a0, bb0, acc);
      acc = wmma_f16(a1, bb1, acc);
      const int c = ct * 16 + N;
      const size_t base = (size_t)c * HW + p0 + M0;   // 32B aligned
      const float4* fp = (const float4*)(fmb + base);
      float4* op = (float4*)(outb + base);
      float4 f0 = fp[0], f1 = fp[1];
      float4 o0 = make_float4(f0.x + acc[0], f0.y + acc[1],
                              f0.z + acc[2], f0.w + acc[3]);
      float4 o1 = make_float4(f1.x + acc[4], f1.y + acc[5],
                              f1.z + acc[6], f1.w + acc[7]);
      op[0] = o0; op[1] = o1;
    }
  }
}

// ---------------------------------------------------------------------------
// Workspace layout (bytes, all offsets 256-aligned)
// ---------------------------------------------------------------------------
constexpr size_t OFF_WS1H  = 0;                         // 64*256 f16 = 32768
constexpr size_t OFF_WT2NH = 32768;                     // 32768
constexpr size_t OFF_WN2TH = 65536;                     // 32768
constexpr size_t OFF_HGH   = 98304;                     // 16*64*64 f16 = 131072
constexpr size_t OFF_HGTH  = 229376;                    // 131072
constexpr size_t OFF_IDX   = 360448;                    // 16*64 i32 = 4096
constexpr size_t OFF_SCORE = 364544;                    // 16*9216 f32 = 589824
constexpr size_t OFF_TOK   = 954368;                    // 16*9216*64 f16 = 18874368

extern "C" void kernel_launch(void* const* d_in, const int* in_sizes, int n_in,
                              void* d_out, int out_size, void* d_ws, size_t ws_size,
                              hipStream_t stream) {
  const float* fmap = (const float*)d_in[0];
  const float* ws1  = (const float*)d_in[1];
  const float* bs1  = (const float*)d_in[2];
  const float* ws2  = (const float*)d_in[3];
  const float* bs2  = (const float*)d_in[4];
  const float* wt2n = (const float*)d_in[5];
  const float* wn2t = (const float*)d_in[6];
  const float* wg1  = (const float*)d_in[7];
  const float* wg2  = (const float*)d_in[8];

  char* ws = (char*)d_ws;
  _Float16* ws1h   = (_Float16*)(ws + OFF_WS1H);
  _Float16* wt2nh  = (_Float16*)(ws + OFF_WT2NH);
  _Float16* wn2th  = (_Float16*)(ws + OFF_WN2TH);
  _Float16* Hgh    = (_Float16*)(ws + OFF_HGH);
  _Float16* HgTh   = (_Float16*)(ws + OFF_HGTH);
  int*      idx    = (int*)     (ws + OFF_IDX);
  float*    scores = (float*)   (ws + OFF_SCORE);
  _Float16* tokens = (_Float16*)(ws + OFF_TOK);

  // K0: weight conversion (layouts already [n][k] row-major)
  ssc_k0_cvt<<<64, 256, 0, stream>>>(ws1,  ws1h,  SHID * CCH);
  ssc_k0_cvt<<<64, 256, 0, stream>>>(wt2n, wt2nh, NDIM * CCH);
  ssc_k0_cvt<<<64, 256, 0, stream>>>(wn2t, wn2th, CCH * NDIM);

  // K1: fused score + token projection (WMMA)
  dim3 g1(NTILE, BATCH);
  ssc_k1_score_proj<<<g1, 128, 0, stream>>>(fmap, ws1h, wt2nh, bs1, ws2, bs2,
                                            tokens, scores);
  // K2: top-64 per batch
  ssc_k2_topk<<<BATCH, 256, 0, stream>>>(scores, idx);

  // K3: adjacency + GCN
  ssc_k3_graph<<<BATCH, 256, 0, stream>>>(tokens, idx, wg1, wg2, Hgh, HgTh);

  // K4: fused attention broadcast + back-projection + residual (WMMA)
  dim3 g4(NTILE, BATCH);
  ssc_k4_attn<<<g4, 128, 0, stream>>>(fmap, tokens, Hgh, HgTh, wn2th,
                                      (float*)d_out);
}
